// GraphLevelAttention_38113539785081
// MI455X (gfx1250) — matrix operand order repaired and checked by reading, not compiled, not run on hardware
//
#include <hip/hip_runtime.h>
#include <hip/hip_bf16.h>

typedef __attribute__((ext_vector_type(16))) _Float16 v16h;
typedef __attribute__((ext_vector_type(8)))  _Float16 v8h;
typedef __attribute__((ext_vector_type(8)))  float    v8f;

#define IN_FEATS 128
#define HEADS 4
#define OUT_FEATS 16
#define HO 64            // HEADS*OUT_FEATS
#define PROJ_W 128       // fused [src(64) | dst(64)] projection width
#define NEG_SLOPE 0.2f

// ---------------------------------------------------------------------------
// K0: convert feat (f32, N x 128) -> featH (f16, N x 128)
// ---------------------------------------------------------------------------
__global__ void k_feat_to_half(const float* __restrict__ feat,
                               _Float16* __restrict__ featH, int total) {
    int idx = blockIdx.x * blockDim.x + threadIdx.x;
    if (idx < total) featH[idx] = (_Float16)feat[idx];
}

// ---------------------------------------------------------------------------
// K1: pack combined weight [128 x 128] (cols 0..63 = W_src, 64..127 = W_dst)
// into WMMA B-fragment layout:
//   Bfrag[((t*4 + ks)*32 + lane)*16 + j]  where
//   n = lane&15, col = t*16+n, K = ks*32 + (lane>>4)*16 + j
// 16384 halfs total.
// ---------------------------------------------------------------------------
__global__ void k_pack_B(const float* __restrict__ W_src,
                         const float* __restrict__ W_dst,
                         _Float16* __restrict__ Bfrag) {
    int idx = blockIdx.x * blockDim.x + threadIdx.x;   // < 16384
    if (idx >= 8 * 4 * 32 * 16) return;
    int j    = idx & 15;
    int lane = (idx >> 4) & 31;
    int ks   = (idx >> 9) & 3;
    int t    = idx >> 11;
    int col  = t * 16 + (lane & 15);
    int K    = ks * 32 + ((lane >> 4) << 4) + j;
    float w  = (col < HO) ? W_src[K * HO + col] : W_dst[K * HO + (col - HO)];
    Bfrag[idx] = (_Float16)w;
}

// ---------------------------------------------------------------------------
// K2: fused projection GEMM: proj[N x 128] = featH[N x 128] @ Wcomb[128 x 128]
// + bias. Block = 256 threads = 8 waves; wave w computes the 16x16 tile at
// column tile w; 4 chained v_wmma_f32_16x16x32_f16 over K=128.
// ---------------------------------------------------------------------------
__global__ __launch_bounds__(256)
void k_gemm_proj(const _Float16* __restrict__ featH,
                 const _Float16* __restrict__ Bfrag,
                 const float* __restrict__ b_src,
                 const float* __restrict__ b_dst,
                 float* __restrict__ proj, int N) {
    const int wave = threadIdx.x >> 5;   // 0..7 -> column tile
    const int lane = threadIdx.x & 31;
    const int row0 = blockIdx.x * 16;
    const int m    = lane & 15;
    const int hi   = lane >> 4;          // K-half selector for A

    // A row pointer (clamped for safety on a ragged last tile; N%16==0 here)
    int arow = row0 + m;
    if (arow >= N) arow = N - 1;
    const _Float16* ap = featH + (size_t)arow * IN_FEATS + hi * 16;

    v8f acc = {};
#pragma unroll
    for (int ks = 0; ks < 4; ++ks) {
        const _Float16* a0 = ap + ks * 32;
        v8h lo  = *(const v8h*)(a0);        // K = ks*32 + hi*16 + 0..7
        v8h hi8 = *(const v8h*)(a0 + 8);    // K = ks*32 + hi*16 + 8..15
        v16h a;
#pragma unroll
        for (int i = 0; i < 8; ++i) { a[i] = lo[i]; a[i + 8] = hi8[i]; }

        v16h b = *(const v16h*)(Bfrag + (((wave * 4 + ks) * 32 + lane) << 4));

        acc = __builtin_amdgcn_wmma_f32_16x16x32_f16(
            /*neg_a=*/false, a, /*neg_b=*/false, b,
            /*c_mod=*/(short)0, acc, /*reuse_a=*/false, /*reuse_b=*/false);
    }

    // Epilogue: lane holds column n = lane&15, rows j + 8*(lane>>4)
    const int col  = wave * 16 + (lane & 15);
    const float bias = (col < HO) ? b_src[col] : b_dst[col - HO];
    float* pout = proj + (size_t)(row0 + 8 * hi) * PROJ_W + col;

    if (row0 + 16 <= N) {
        // full tile: single branch, unguarded store clause
#pragma unroll
        for (int j = 0; j < 8; ++j) pout[(size_t)j * PROJ_W] = acc[j] + bias;
    } else {
        // ragged last tile
#pragma unroll
        for (int j = 0; j < 8; ++j) {
            if (row0 + 8 * hi + j < N) pout[(size_t)j * PROJ_W] = acc[j] + bias;
        }
    }
}

// ---------------------------------------------------------------------------
// K3: init seg_max = -inf, seg_sum = 0, out = 0
// ---------------------------------------------------------------------------
__global__ void k_init(float* __restrict__ seg_max, float* __restrict__ seg_sum,
                       float* __restrict__ out, int nh, int nho) {
    int idx = blockIdx.x * blockDim.x + threadIdx.x;
    if (idx < nho) out[idx] = 0.0f;
    if (idx < nh) {
        seg_max[idx] = -__builtin_huge_valf();
        seg_sum[idx] = 0.0f;
    }
}

__device__ __forceinline__ void atomicMaxF(float* addr, float v) {
    // IEEE bit trick: valid with -inf init; positives via signed max,
    // negatives via unsigned min.
    if (v >= 0.0f) atomicMax((int*)addr, __float_as_int(v));
    else           atomicMin((unsigned int*)addr, __float_as_uint(v));
}

// ---------------------------------------------------------------------------
// K4: per (edge, head): logit = sum_o attn[h][o]*leaky_relu(el+er);
//     store logit; atomic segment max over dst.
// ---------------------------------------------------------------------------
__global__ void k_edge_logits(const float* __restrict__ proj,
                              const float* __restrict__ attn,
                              const int* __restrict__ src,
                              const int* __restrict__ dst,
                              float* __restrict__ logits,
                              float* __restrict__ seg_max, int EH) {
    int idx = blockIdx.x * blockDim.x + threadIdx.x;
    if (idx >= EH) return;
    int e = idx >> 2;
    int h = idx & 3;
    int s = src[e];
    int d = dst[e];
    const float4* el = (const float4*)(proj + (size_t)s * PROJ_W + h * OUT_FEATS);
    const float4* er = (const float4*)(proj + (size_t)d * PROJ_W + HO + h * OUT_FEATS);
    const float4* at = (const float4*)(attn + h * OUT_FEATS);
    float logit = 0.0f;
#pragma unroll
    for (int j = 0; j < 4; ++j) {
        float4 a = el[j], b = er[j], w = at[j];
        float x;
        x = a.x + b.x; x = x > 0.0f ? x : NEG_SLOPE * x; logit += w.x * x;
        x = a.y + b.y; x = x > 0.0f ? x : NEG_SLOPE * x; logit += w.y * x;
        x = a.z + b.z; x = x > 0.0f ? x : NEG_SLOPE * x; logit += w.z * x;
        x = a.w + b.w; x = x > 0.0f ? x : NEG_SLOPE * x; logit += w.w * x;
    }
    logits[idx] = logit;
    atomicMaxF(&seg_max[d * HEADS + h], logit);
}

// ---------------------------------------------------------------------------
// K5: ex = exp(logit - seg_max[dst]); overwrite logits with ex; atomic segsum.
// ---------------------------------------------------------------------------
__global__ void k_edge_exp(const int* __restrict__ dst,
                           float* __restrict__ logits,
                           const float* __restrict__ seg_max,
                           float* __restrict__ seg_sum, int EH) {
    int idx = blockIdx.x * blockDim.x + threadIdx.x;
    if (idx >= EH) return;
    int e = idx >> 2;
    int h = idx & 3;
    int d = dst[e];
    float ex = __expf(logits[idx] - seg_max[d * HEADS + h]);
    logits[idx] = ex;
    atomicAdd(&seg_sum[d * HEADS + h], ex);
}

// ---------------------------------------------------------------------------
// K6: a = ex / seg_sum[dst]; out[dst] += el * a (16 f32 atomics per edge-head)
// ---------------------------------------------------------------------------
__global__ void k_edge_scatter(const float* __restrict__ proj,
                               const int* __restrict__ src,
                               const int* __restrict__ dst,
                               const float* __restrict__ logits,
                               const float* __restrict__ seg_sum,
                               float* __restrict__ out, int EH) {
    int idx = blockIdx.x * blockDim.x + threadIdx.x;
    if (idx >= EH) return;
    int e = idx >> 2;
    int h = idx & 3;
    int s = src[e];
    int d = dst[e];
    float a = logits[idx] / seg_sum[d * HEADS + h];
    const float* el = proj + (size_t)s * PROJ_W + h * OUT_FEATS;
    float* op = out + (size_t)d * HO + h * OUT_FEATS;
#pragma unroll
    for (int o = 0; o < OUT_FEATS; ++o) atomicAdd(&op[o], el[o] * a);
}

// ---------------------------------------------------------------------------
extern "C" void kernel_launch(void* const* d_in, const int* in_sizes, int n_in,
                              void* d_out, int out_size, void* d_ws, size_t ws_size,
                              hipStream_t stream) {
    const float* feat  = (const float*)d_in[0];
    const float* W_src = (const float*)d_in[1];
    const float* b_src = (const float*)d_in[2];
    const float* W_dst = (const float*)d_in[3];
    const float* b_dst = (const float*)d_in[4];
    const float* attn  = (const float*)d_in[5];
    const int*   src   = (const int*)d_in[6];
    const int*   dst   = (const int*)d_in[7];
    float* out = (float*)d_out;

    const int N = in_sizes[0] / IN_FEATS;   // 50000
    const int E = in_sizes[6];              // 800000
    const int EH = E * HEADS;

    // workspace carve-up (all 256B aligned)
    char* ws = (char*)d_ws;
    size_t off = 0;
    _Float16* featH = (_Float16*)(ws + off); off += (size_t)N * IN_FEATS * 2;   // 12.8 MB
    off = (off + 255) & ~(size_t)255;
    _Float16* Bfrag = (_Float16*)(ws + off); off += 8 * 4 * 32 * 16 * 2;        // 32 KB
    off = (off + 255) & ~(size_t)255;
    float* proj     = (float*)(ws + off);    off += (size_t)N * PROJ_W * 4;     // 25.6 MB
    off = (off + 255) & ~(size_t)255;
    float* logits   = (float*)(ws + off);    off += (size_t)EH * 4;             // 12.8 MB
    off = (off + 255) & ~(size_t)255;
    float* seg_max  = (float*)(ws + off);    off += (size_t)N * HEADS * 4;      // 0.8 MB
    off = (off + 255) & ~(size_t)255;
    float* seg_sum  = (float*)(ws + off);    off += (size_t)N * HEADS * 4;      // 0.8 MB
    (void)ws_size;

    const int TB = 256;
    int featTot = N * IN_FEATS;

    k_feat_to_half<<<(featTot + TB - 1) / TB, TB, 0, stream>>>(feat, featH, featTot);
    k_pack_B<<<(8 * 4 * 32 * 16 + TB - 1) / TB, TB, 0, stream>>>(W_src, W_dst, Bfrag);
    k_gemm_proj<<<(N + 15) / 16, TB, 0, stream>>>(featH, Bfrag, b_src, b_dst, proj, N);

    int nh = N * HEADS, nho = N * HO;
    k_init<<<(nho + TB - 1) / TB, TB, 0, stream>>>(seg_max, seg_sum, out, nh, nho);

    int egrid = (EH + TB - 1) / TB;
    k_edge_logits<<<egrid, TB, 0, stream>>>(proj, attn, src, dst, logits, seg_max, EH);
    k_edge_exp<<<egrid, TB, 0, stream>>>(dst, logits, seg_max, seg_sum, EH);
    k_edge_scatter<<<egrid, TB, 0, stream>>>(proj, src, dst, logits, seg_sum, out, EH);
}
